// CustomTransformerEncoderLayer_44676249813047
// MI455X (gfx1250) — matrix operand order repaired and checked by reading, not compiled, not run on hardware
//
#include <hip/hip_runtime.h>

#define DEV __device__ __forceinline__

typedef __attribute__((ext_vector_type(16))) __bf16 v16bf;
typedef __attribute__((ext_vector_type(8)))  __bf16 v8bf;
typedef __attribute__((ext_vector_type(8)))  float  v8f;

static constexpr int Bn = 2, Sn = 8192, Dn = 768, Hn = 12, HDn = 64, Wn = 256, DFFn = 3072;
static constexpr int Mn = Bn * Sn; // 16384 rows

// ---------- small helpers ----------
DEV v8bf zero8() {
  v8bf z;
  #pragma unroll
  for (int i = 0; i < 8; ++i) z[i] = (__bf16)0.f;
  return z;
}
DEV v8f zf8() {
  v8f z;
  #pragma unroll
  for (int i = 0; i < 8; ++i) z[i] = 0.f;
  return z;
}
DEV v16bf cat8(v8bf lo, v8bf hi) {
  return __builtin_shufflevector(lo, hi, 0,1,2,3,4,5,6,7,8,9,10,11,12,13,14,15);
}
DEV v8bf ld8(const __bf16* p) { return *(const v8bf*)p; }
DEV v8f wmma_bf16(v16bf a, v16bf b, v8f c) {
  // D = A(16x32 bf16) * B(32x16 bf16) + C(16x16 f32)
  return __builtin_amdgcn_wmma_f32_16x16x32_bf16(false, a, false, b, (short)0, c, false, false);
}

// CDNA5 async global->LDS copy (16B per lane), tracked by ASYNCcnt
DEV void async_ld16(const __bf16* lds_dst, const __bf16* gsrc) {
  const unsigned ldsa = (unsigned)(uintptr_t)lds_dst;  // low 32 bits = LDS byte address
  asm volatile("global_load_async_to_lds_b128 %0, %1, off"
               :: "v"(ldsa), "v"(gsrc) : "memory");
}
DEV void wait_async0() { asm volatile("s_wait_asynccnt 0x0" ::: "memory"); }

// block-wide sum over 256 threads (8 waves of 32)
DEV float blk_sum(float val, float* sm) {
  #pragma unroll
  for (int o = 16; o > 0; o >>= 1) val += __shfl_xor(val, o, 32);
  const int wv = threadIdx.x >> 5;
  __syncthreads();
  if ((threadIdx.x & 31) == 0) sm[wv] = val;
  __syncthreads();
  float s = 0.f;
  #pragma unroll
  for (int i = 0; i < 8; ++i) s += sm[i];
  return s;
}

// ---------- weight prep: f32 [K][N] -> bf16 [N][K] ----------
__global__ __launch_bounds__(256) void transpose_bf16_kernel(
    const float* __restrict__ w, __bf16* __restrict__ wt, int K, int N) {
  size_t id = (size_t)blockIdx.x * 256 + threadIdx.x;
  if (id < (size_t)K * N) {
    int kk = (int)(id / N), n = (int)(id % N);
    wt[(size_t)n * K + kk] = (__bf16)w[id];
  }
}

// ---------- fused LN kernels (one row of D=768 per block, 3 elems/thread) ----------
__global__ __launch_bounds__(256) void ln_embed_kernel(
    const float* __restrict__ src, const float* __restrict__ pos, const float* __restrict__ tt,
    const float* __restrict__ n1s, const float* __restrict__ n1b,
    const float* __restrict__ es, const float* __restrict__ eb,
    float* __restrict__ hf, __bf16* __restrict__ hb) {
  __shared__ float sm[8];
  const int row = blockIdx.x, tid = threadIdx.x;
  const int srow = row % Sn;
  const float* xr = src + (size_t)row * Dn;
  float x[3], u[3];
  float s1 = 0.f, q1 = 0.f;
  #pragma unroll
  for (int i = 0; i < 3; ++i) { int j = i * 256 + tid; x[i] = xr[j]; s1 += x[i]; q1 += x[i] * x[i]; }
  s1 = blk_sum(s1, sm); q1 = blk_sum(q1, sm);
  float mu = s1 * (1.f / 768.f);
  float rs = rsqrtf(q1 * (1.f / 768.f) - mu * mu + 1e-5f);
  float s2 = 0.f, q2 = 0.f;
  #pragma unroll
  for (int i = 0; i < 3; ++i) {
    int j = i * 256 + tid;
    float t = (x[i] - mu) * rs * n1s[j] + n1b[j];
    float uu = t + pos[(size_t)srow * Dn + j] + tt[j];
    u[i] = uu; s2 += uu; q2 += uu * uu;
  }
  s2 = blk_sum(s2, sm); q2 = blk_sum(q2, sm);
  float mu2 = s2 * (1.f / 768.f);
  float rs2 = rsqrtf(q2 * (1.f / 768.f) - mu2 * mu2 + 1e-12f);
  #pragma unroll
  for (int i = 0; i < 3; ++i) {
    int j = i * 256 + tid;
    float hv = (u[i] - mu2) * rs2 * es[j] + eb[j];
    hf[(size_t)row * Dn + j] = hv;
    hb[(size_t)row * Dn + j] = (__bf16)hv;
  }
}

__global__ __launch_bounds__(256) void ln_res_kernel(
    const float* __restrict__ A, const float* __restrict__ Bv,
    const float* __restrict__ sc, const float* __restrict__ bi,
    float eps, float* __restrict__ of, __bf16* __restrict__ ob) {
  __shared__ float sm[8];
  const int row = blockIdx.x, tid = threadIdx.x;
  float x[3];
  float s = 0.f, q = 0.f;
  #pragma unroll
  for (int i = 0; i < 3; ++i) {
    int j = i * 256 + tid;
    x[i] = A[(size_t)row * Dn + j] + Bv[(size_t)row * Dn + j];
    s += x[i]; q += x[i] * x[i];
  }
  s = blk_sum(s, sm); q = blk_sum(q, sm);
  float mu = s * (1.f / 768.f);
  float rs = rsqrtf(q * (1.f / 768.f) - mu * mu + eps);
  #pragma unroll
  for (int i = 0; i < 3; ++i) {
    int j = i * 256 + tid;
    float hv = (x[i] - mu) * rs * sc[j] + bi[j];
    of[(size_t)row * Dn + j] = hv;
    ob[(size_t)row * Dn + j] = (__bf16)hv;
  }
}

// o = ln(h2+ffo, out_ln, 1e-12); src2 = src + o; g = ln(src2, n2, 1e-5)
__global__ __launch_bounds__(256) void ln_final_kernel(
    const float* __restrict__ h2f, const float* __restrict__ ffo, const float* __restrict__ src,
    const float* __restrict__ ols, const float* __restrict__ olb,
    const float* __restrict__ n2s, const float* __restrict__ n2b,
    float* __restrict__ src2, __bf16* __restrict__ gb) {
  __shared__ float sm[8];
  const int row = blockIdx.x, tid = threadIdx.x;
  float x[3], u[3];
  float s1 = 0.f, q1 = 0.f;
  #pragma unroll
  for (int i = 0; i < 3; ++i) {
    int j = i * 256 + tid;
    x[i] = h2f[(size_t)row * Dn + j] + ffo[(size_t)row * Dn + j];
    s1 += x[i]; q1 += x[i] * x[i];
  }
  s1 = blk_sum(s1, sm); q1 = blk_sum(q1, sm);
  float mu = s1 * (1.f / 768.f);
  float rs = rsqrtf(q1 * (1.f / 768.f) - mu * mu + 1e-12f);
  float s2 = 0.f, q2 = 0.f;
  #pragma unroll
  for (int i = 0; i < 3; ++i) {
    int j = i * 256 + tid;
    float o = (x[i] - mu) * rs * ols[j] + olb[j];
    float s2v = src[(size_t)row * Dn + j] + o;
    src2[(size_t)row * Dn + j] = s2v;
    u[i] = s2v; s2 += s2v; q2 += s2v * s2v;
  }
  s2 = blk_sum(s2, sm); q2 = blk_sum(q2, sm);
  float mu2 = s2 * (1.f / 768.f);
  float rs2 = rsqrtf(q2 * (1.f / 768.f) - mu2 * mu2 + 1e-5f);
  #pragma unroll
  for (int i = 0; i < 3; ++i) {
    int j = i * 256 + tid;
    gb[(size_t)row * Dn + j] = (__bf16)((u[i] - mu2) * rs2 * n2s[j] + n2b[j]);
  }
}

// ---------- WMMA GEMM: C[M,N] = act((A[M,K]*Wt[N,K]^T + bias) * scale) (+ Res) ----------
// Block tile 128x128, 8 waves of 32x64 each, K step 32.
// Double-buffered LDS staging via CDNA5 global_load_async_to_lds_b128 (ASYNCcnt).
template <int ACT, bool WF32, bool WBF16, bool RES>
__global__ __launch_bounds__(256) void gemm_bf16_kernel(
    const __bf16* __restrict__ A, const __bf16* __restrict__ Wt,
    const float* __restrict__ bias, float* __restrict__ Cf, __bf16* __restrict__ Cb,
    const float* __restrict__ Res, int M, int N, int K, float scale) {
  __shared__ __align__(16) __bf16 lA[2][128 * 40];
  __shared__ __align__(16) __bf16 lB[2][128 * 40];
  const int tid = threadIdx.x, lane = tid & 31, wave = tid >> 5;
  const int l16 = lane & 15, lh = lane >> 4;
  const int m0 = blockIdx.y * 128, n0 = blockIdx.x * 128;
  const int wm = (wave & 3) * 32, wn = (wave >> 2) * 64;
  // this thread's staging slots: row r, two 16B segments at cb and cb+8
  const int r = tid >> 1, cb = (tid & 1) * 16;
  v8f acc[2][4];
  #pragma unroll
  for (int i = 0; i < 2; ++i)
    #pragma unroll
    for (int j = 0; j < 4; ++j) acc[i][j] = zf8();

  // issue async copies of the K-slab at k0 into buffer `buf`
  auto issue = [&](int buf, int k0) {
    async_ld16(&lA[buf][r * 40 + cb],     &A[(size_t)(m0 + r) * K + k0 + cb]);
    async_ld16(&lA[buf][r * 40 + cb + 8], &A[(size_t)(m0 + r) * K + k0 + cb + 8]);
    async_ld16(&lB[buf][r * 40 + cb],     &Wt[(size_t)(n0 + r) * K + k0 + cb]);
    async_ld16(&lB[buf][r * 40 + cb + 8], &Wt[(size_t)(n0 + r) * K + k0 + cb + 8]);
  };

  issue(0, 0);
  for (int k0 = 0; k0 < K; k0 += 32) {
    const int buf = (k0 >> 5) & 1;
    wait_async0();     // data for `buf` landed in LDS
    __syncthreads();   // visible to all waves; previous compute done
    if (k0 + 32 < K) issue(buf ^ 1, k0 + 32);  // overlap next copy with compute

    // A fragment: lane<16 holds K 0..7 / 16..23, lane>=16 holds K 8..15 / 24..31
    const int akb = lh * 8;
    // B fragment: lane<16 holds K 0..7 / 16..23 at N=l16, lane>=16 holds K 8..15 / 24..31
    const int bkb = lh * 8;
    v16bf af[2], bfr[4];
    #pragma unroll
    for (int mt = 0; mt < 2; ++mt) {
      const __bf16* p = &lA[buf][(wm + mt * 16 + l16) * 40 + akb];
      af[mt] = cat8(ld8(p), ld8(p + 16));
    }
    #pragma unroll
    for (int nt = 0; nt < 4; ++nt) {
      const __bf16* p = &lB[buf][(wn + nt * 16 + l16) * 40 + bkb];
      bfr[nt] = cat8(ld8(p), ld8(p + 16));
    }
    #pragma unroll
    for (int mt = 0; mt < 2; ++mt)
      #pragma unroll
      for (int nt = 0; nt < 4; ++nt)
        acc[mt][nt] = wmma_bf16(af[mt], bfr[nt], acc[mt][nt]);
  }

  // epilogue: C layout = VGPR v, lanes 0-15: row v, col lane; lanes 16-31: row v+8
  #pragma unroll
  for (int mt = 0; mt < 2; ++mt)
    for (int nt = 0; nt < 4; ++nt) {
      const int col = n0 + wn + nt * 16 + l16;
      const float bz = bias[col];
      #pragma unroll
      for (int v = 0; v < 8; ++v) {
        const int row = m0 + wm + mt * 16 + v + lh * 8;
        float x = (acc[mt][nt][v] + bz) * scale;
        if (ACT == 1) x = 0.5f * x * (1.f + erff(x * 0.70710678118f));  // exact GELU
        if (ACT == 2) x = fmaxf(x, 0.f);                                // ReLU
        const size_t idx = (size_t)row * N + col;
        if (RES) x += Res[idx];
        if (WF32) Cf[idx] = x;
        if (WBF16) Cb[idx] = (__bf16)x;
      }
    }
}

// ---------- banded flash attention: one block per (chunk, head, batch) ----------
// 512 threads = 16 waves, each wave owns 16 query rows -> ~half the VGPR pressure
// of a 32-row wave tile (keeps allocation under 256 VGPRs, no MSB toggling).
// Q chunk staged in LDS; interior key blocks skip masking via wave-uniform branch.
__global__ __launch_bounds__(512) void band_attn_kernel(
    const __bf16* __restrict__ q, const __bf16* __restrict__ k,
    const __bf16* __restrict__ v, __bf16* __restrict__ a) {
  __shared__ __align__(16) __bf16 ql[256 * 72];      // Q chunk: [row][dim], padded
  __shared__ __align__(16) __bf16 vt[64 * 72];       // V^T block: [dim][key], padded
  __shared__ __align__(16) __bf16 pl[16 * 16 * 72];  // per-wave P re-layout buffer
  const int tid = threadIdx.x, lane = tid & 31, wave = tid >> 5;
  const int l16 = lane & 15, lh = lane >> 4;
  const int c = blockIdx.x, h = blockIdx.y, b = blockIdx.z;
  const int qrow0 = c * Wn + wave * 16;
  const size_t bsd = (size_t)b * Sn;

  // stage Q chunk once: thread -> half a query row (32 dims)
  {
    const int qrw = tid >> 1, dbase = (tid & 1) * 32;
    const __bf16* p = &q[(bsd + c * Wn + qrw) * Dn + h * HDn + dbase];
    #pragma unroll
    for (int s8 = 0; s8 < 4; ++s8)
      *(v8bf*)&ql[qrw * 72 + dbase + s8 * 8] = ld8(p + s8 * 8);
  }

  v8f oacc[4];
  #pragma unroll
  for (int j = 0; j < 4; ++j) oacc[j] = zf8();
  float rmax[8], rsum[8];
  #pragma unroll
  for (int vv = 0; vv < 8; ++vv) { rmax[vv] = -3.0e38f; rsum[vv] = 0.f; }

  for (int kb = 0; kb < 12; ++kb) {
    const int key0 = c * Wn - Wn + kb * 64;
    __syncthreads();
    { // stage V^T block: thread -> one key row, 8 dims
      const int key = tid >> 3, dseg = (tid & 7) * 8;
      const int keyr = key0 + key;
      v8bf d0 = zero8();
      if (keyr >= 0 && keyr < Sn)
        d0 = ld8(&v[(bsd + keyr) * Dn + h * HDn + dseg]);
      #pragma unroll
      for (int e = 0; e < 8; ++e)
        vt[(dseg + e) * 72 + key] = d0[e];
    }
    __syncthreads();

    // scores = Q * K^T (B-fragment = contiguous head dims of K row)
    v8f sacc[4];
    #pragma unroll
    for (int j = 0; j < 4; ++j) sacc[j] = zf8();
    #pragma unroll
    for (int ks = 0; ks < 2; ++ks) {
      v16bf bfr[4];
      #pragma unroll
      for (int nt = 0; nt < 4; ++nt) {
        const int keyr = key0 + nt * 16 + l16;
        if (keyr >= 0 && keyr < Sn) {
          const __bf16* p = &k[(bsd + keyr) * Dn + h * HDn + lh * 8 + ks * 32];
          bfr[nt] = cat8(ld8(p), ld8(p + 16));
        } else {
          bfr[nt] = cat8(zero8(), zero8());
        }
      }
      const __bf16* p = &ql[(wave * 16 + l16) * 72 + lh * 8 + ks * 32];
      const v16bf af = cat8(ld8(p), ld8(p + 16));
      #pragma unroll
      for (int nt = 0; nt < 4; ++nt)
        sacc[nt] = wmma_bf16(af, bfr[nt], sacc[nt]);
    }

    // wave-uniform: is this key block fully valid (in-range and inside the band)?
    const bool full = (key0 >= 0) && (key0 + 64 <= Sn) &&
                      ((key0 + 63 - qrow0) <= Wn) && ((qrow0 + 15 - key0) <= Wn);

    // band mask + online softmax (row stats per v slot; 16-lane reductions)
    auto softmax_step = [&](bool MASK) {
      #pragma unroll
      for (int vv = 0; vv < 8; ++vv) {
        const int qg = qrow0 + vv + lh * 8;
        float bm = -3.0e38f;
        #pragma unroll
        for (int nt = 0; nt < 4; ++nt) {
          float x = sacc[nt][vv];
          if (MASK) {
            const int kp = key0 + nt * 16 + l16;
            const int df = kp - qg;
            const bool ok = (kp >= 0) && (kp < Sn) && (df <= Wn) && (df >= -Wn);
            x = ok ? x : -1.0e9f;
            sacc[nt][vv] = x;
          }
          bm = fmaxf(bm, x);
        }
        #pragma unroll
        for (int o = 1; o < 16; o <<= 1) bm = fmaxf(bm, __shfl_xor(bm, o, 32));
        const float nm = fmaxf(rmax[vv], bm);
        const float corr = __expf(rmax[vv] - nm);
        float ps = 0.f;
        #pragma unroll
        for (int nt = 0; nt < 4; ++nt) {
          const float p = __expf(sacc[nt][vv] - nm);
          sacc[nt][vv] = p;
          ps += p;
        }
        #pragma unroll
        for (int o = 1; o < 16; o <<= 1) ps += __shfl_xor(ps, o, 32);
        rsum[vv] = rsum[vv] * corr + ps;
        rmax[vv] = nm;
        #pragma unroll
        for (int nt = 0; nt < 4; ++nt) oacc[nt][vv] *= corr;
      }
    };
    if (full) softmax_step(false);
    else      softmax_step(true);

    // C-layout -> A-layout re-shape of P through per-wave LDS
    __bf16* pw = &pl[wave * 16 * 72];
    #pragma unroll
    for (int nt = 0; nt < 4; ++nt)
      #pragma unroll
      for (int vv = 0; vv < 8; ++vv)
        pw[(vv + lh * 8) * 72 + nt * 16 + l16] = (__bf16)sacc[nt][vv];

    // O += P * V
    #pragma unroll
    for (int ks = 0; ks < 2; ++ks) {
      const __bf16* pp = &pw[l16 * 72 + lh * 8 + ks * 32];
      const v16bf pf = cat8(ld8(pp), ld8(pp + 16));
      v16bf vf[4];
      #pragma unroll
      for (int nt = 0; nt < 4; ++nt) {
        const __bf16* p = &vt[(nt * 16 + l16) * 72 + lh * 8 + ks * 32];
        vf[nt] = cat8(ld8(p), ld8(p + 16));
      }
      #pragma unroll
      for (int nt = 0; nt < 4; ++nt)
        oacc[nt] = wmma_bf16(pf, vf[nt], oacc[nt]);
    }
  }

  // normalize and write back in [B,S,D] (heads merged) layout
  #pragma unroll
  for (int nt = 0; nt < 4; ++nt)
    #pragma unroll
    for (int vv = 0; vv < 8; ++vv) {
      const int row = qrow0 + vv + lh * 8;
      const float val = oacc[nt][vv] / rsum[vv];
      a[(bsd + row) * Dn + h * HDn + nt * 16 + l16] = (__bf16)val;
    }
}

// ---------- host orchestration ----------
extern "C" void kernel_launch(void* const* d_in, const int* in_sizes, int n_in,
                              void* d_out, int out_size, void* d_ws, size_t ws_size,
                              hipStream_t stream) {
  (void)in_sizes; (void)n_in; (void)out_size; (void)ws_size;
  const float* src = (const float*)d_in[0];
  const float* pos = (const float*)d_in[1];
  const float* tt  = (const float*)d_in[2];
  const float* els = (const float*)d_in[3];
  const float* elb = (const float*)d_in[4];
  const float* wq  = (const float*)d_in[5];
  const float* bq  = (const float*)d_in[6];
  const float* wk  = (const float*)d_in[7];
  const float* bk  = (const float*)d_in[8];
  const float* wv  = (const float*)d_in[9];
  const float* bv  = (const float*)d_in[10];
  const float* wo  = (const float*)d_in[11];
  const float* bo  = (const float*)d_in[12];
  const float* als = (const float*)d_in[13];
  const float* alb = (const float*)d_in[14];
  const float* wi  = (const float*)d_in[15];
  const float* bi  = (const float*)d_in[16];
  const float* wo2 = (const float*)d_in[17];
  const float* bo2 = (const float*)d_in[18];
  const float* ols = (const float*)d_in[19];
  const float* olb = (const float*)d_in[20];
  const float* n1s = (const float*)d_in[21];
  const float* n1b = (const float*)d_in[22];
  const float* n2s = (const float*)d_in[23];
  const float* n2b = (const float*)d_in[24];
  const float* w1  = (const float*)d_in[25];
  const float* b1  = (const float*)d_in[26];
  const float* w2  = (const float*)d_in[27];
  const float* b2  = (const float*)d_in[28];
  float* out = (float*)d_out;

  char* base = (char*)d_ws;
  size_t off = 0;
  auto alloc = [&](size_t sz) -> char* {
    char* p = base + off;
    off = (off + sz + 255) & ~(size_t)255;
    return p;
  };
  const size_t MDf = (size_t)Mn * Dn * sizeof(float);
  const size_t MDb = (size_t)Mn * Dn * sizeof(__bf16);
  const size_t MFb = (size_t)Mn * DFFn * sizeof(__bf16);

  float*  h_f   = (float*)alloc(MDf);
  float*  h2_f  = (float*)alloc(MDf);
  float*  src2f = (float*)alloc(MDf);
  float*  tmpf  = (float*)alloc(MDf);
  __bf16* h_b   = (__bf16*)alloc(MDb);
  __bf16* q_b   = (__bf16*)alloc(MDb);
  __bf16* k_b   = (__bf16*)alloc(MDb);
  __bf16* v_b   = (__bf16*)alloc(MDb);
  __bf16* a_b   = (__bf16*)alloc(MDb);
  __bf16* h2_b  = (__bf16*)alloc(MDb);
  __bf16* g_b   = (__bf16*)alloc(MDb);
  __bf16* big_b = (__bf16*)alloc(MFb);
  __bf16* wq_t  = (__bf16*)alloc((size_t)Dn * Dn * 2);
  __bf16* wk_t  = (__bf16*)alloc((size_t)Dn * Dn * 2);
  __bf16* wv_t  = (__bf16*)alloc((size_t)Dn * Dn * 2);
  __bf16* wo_t  = (__bf16*)alloc((size_t)Dn * Dn * 2);
  __bf16* wi_t  = (__bf16*)alloc((size_t)Dn * DFFn * 2);
  __bf16* wo2_t = (__bf16*)alloc((size_t)Dn * DFFn * 2);
  __bf16* w1_t  = (__bf16*)alloc((size_t)Dn * DFFn * 2);
  __bf16* w2_t  = (__bf16*)alloc((size_t)Dn * DFFn * 2);

  const dim3 tb(256);
  const int nsmall = Dn * Dn, nbig = Dn * DFFn;
  transpose_bf16_kernel<<<(nsmall + 255) / 256, tb, 0, stream>>>(wq, wq_t, Dn, Dn);
  transpose_bf16_kernel<<<(nsmall + 255) / 256, tb, 0, stream>>>(wk, wk_t, Dn, Dn);
  transpose_bf16_kernel<<<(nsmall + 255) / 256, tb, 0, stream>>>(wv, wv_t, Dn, Dn);
  transpose_bf16_kernel<<<(nsmall + 255) / 256, tb, 0, stream>>>(wo, wo_t, Dn, Dn);
  transpose_bf16_kernel<<<(nbig + 255) / 256, tb, 0, stream>>>(wi, wi_t, Dn, DFFn);
  transpose_bf16_kernel<<<(nbig + 255) / 256, tb, 0, stream>>>(wo2, wo2_t, DFFn, Dn);
  transpose_bf16_kernel<<<(nbig + 255) / 256, tb, 0, stream>>>(w1, w1_t, Dn, DFFn);
  transpose_bf16_kernel<<<(nbig + 255) / 256, tb, 0, stream>>>(w2, w2_t, DFFn, Dn);

  // h = ln(ln(src,n1)+pos+tt, emb_ln)
  ln_embed_kernel<<<Mn, tb, 0, stream>>>(src, pos, tt, n1s, n1b, els, elb, h_f, h_b);

  const dim3 gqkv(Dn / 128, Mn / 128);   // 6 x 128
  const dim3 gff(DFFn / 128, Mn / 128);  // 24 x 128
  gemm_bf16_kernel<0, false, true, false><<<gqkv, tb, 0, stream>>>(h_b, wq_t, bq, nullptr, q_b, nullptr, Mn, Dn, Dn, 0.125f);
  gemm_bf16_kernel<0, false, true, false><<<gqkv, tb, 0, stream>>>(h_b, wk_t, bk, nullptr, k_b, nullptr, Mn, Dn, Dn, 1.f);
  gemm_bf16_kernel<0, false, true, false><<<gqkv, tb, 0, stream>>>(h_b, wv_t, bv, nullptr, v_b, nullptr, Mn, Dn, Dn, 1.f);

  const dim3 ga(Sn / Wn, Hn, Bn);  // 32 x 12 x 2
  band_attn_kernel<<<ga, dim3(512), 0, stream>>>(q_b, k_b, v_b, a_b);

  // h2 = ln(h + a@wo + bo)
  gemm_bf16_kernel<0, true, false, false><<<gqkv, tb, 0, stream>>>(a_b, wo_t, bo, tmpf, nullptr, nullptr, Mn, Dn, Dn, 1.f);
  ln_res_kernel<<<Mn, tb, 0, stream>>>(h_f, tmpf, als, alb, 1e-12f, h2_f, h2_b);

  // FFN: inter = gelu(h2@wi+bi); ffo = inter@wo2+bo2
  gemm_bf16_kernel<1, false, true, false><<<gff, tb, 0, stream>>>(h2_b, wi_t, bi, nullptr, big_b, nullptr, Mn, DFFn, Dn, 1.f);
  gemm_bf16_kernel<0, true, false, false><<<gqkv, tb, 0, stream>>>(big_b, wo2_t, bo2, tmpf, nullptr, nullptr, Mn, Dn, DFFn, 1.f);

  // src2 = src + ln(h2+ffo, out_ln); g = ln(src2, n2)
  ln_final_kernel<<<Mn, tb, 0, stream>>>(h2_f, tmpf, src, ols, olb, n2s, n2b, src2f, g_b);

  // ff = relu(g@w1+b1)@w2+b2; out = src2 + ff
  gemm_bf16_kernel<2, false, true, false><<<gff, tb, 0, stream>>>(g_b, w1_t, b1, nullptr, big_b, nullptr, Mn, DFFn, Dn, 1.f);
  gemm_bf16_kernel<0, true, false, true><<<gqkv, tb, 0, stream>>>(big_b, w2_t, b2, out, nullptr, src2f, Mn, Dn, DFFn, 1.f);
}
